// CausalSelfAttention_45071386804892
// MI455X (gfx1250) — matrix operand order repaired and checked by reading, not compiled, not run on hardware
//
#include <hip/hip_runtime.h>
#include <hip/hip_bf16.h>
#include <math.h>

// ---------------------------------------------------------------------------
// CausalSelfAttention for MI455X (gfx1250, wave32, WMMA bf16)
//   B=2, S=2048, D_MODEL=2048, NH=32, NKV=8, HD=64, theta=10000
// bf16 WMMA (f32 accumulate) everywhere; flash-attention streaming softmax.
// Weights transpose-converted to bf16 once (W^T[n][k]) -> in-loop staging is
// one contiguous b128 load + b128 DS store; x pre-converted to bf16.
// GEMMs: 32x64 tile per wave, double-buffered LDS B tile, software-pipelined
// global loads, one barrier per k-step, 8 WMMAs per k-step per wave.
// V stored dim-major so attention B-frags are contiguous global loads.
// ---------------------------------------------------------------------------

#define S_LEN   2048
#define DM      2048
#define HD      64
#define NH      32
#define NKV     8

typedef __bf16 v16bf __attribute__((ext_vector_type(16)));
typedef __bf16 bh8   __attribute__((ext_vector_type(8)));
typedef float  v8f   __attribute__((ext_vector_type(8)));
typedef float  fx4   __attribute__((ext_vector_type(4)));

union BF16x16 { v16bf v; bh8 h[2]; };

static __device__ __forceinline__ v8f wmma_bf16(v16bf a, v16bf b, v8f c) {
  // D = A(16x32) * B(32x16) + C, fp32 accumulate
  return __builtin_amdgcn_wmma_f32_16x16x32_bf16(false, a, false, b,
                                                 (short)0, c, false, false);
}

// ---------------------------------------------------------------- utilities
__global__ void cvt_f32_bf16(const float* __restrict__ src,
                             __bf16* __restrict__ dst, int n) {
  int i = blockIdx.x * blockDim.x + threadIdx.x;
  if (i < n) dst[i] = (__bf16)src[i];
}

// W (2048 x N) fp32 -> WT (N x 2048) bf16, 32x32 tiles through LDS, coalesced.
__global__ void __launch_bounds__(256)
transpose_cvt(const float* __restrict__ W, __bf16* __restrict__ WT, int N) {
  __shared__ float tile[32][33];
  const int tx = threadIdx.x & 31, ty = threadIdx.x >> 5;   // 32 x 8
  const int n0 = blockIdx.x * 32, k0 = blockIdx.y * 32;
  #pragma unroll
  for (int r = 0; r < 4; ++r)
    tile[ty + 8 * r][tx] = W[(size_t)(k0 + ty + 8 * r) * N + n0 + tx];
  __syncthreads();
  #pragma unroll
  for (int r = 0; r < 4; ++r)
    WT[(size_t)(n0 + ty + 8 * r) * DM + k0 + tx] = (__bf16)tile[tx][ty + 8 * r];
}

__global__ void rope_table(float* __restrict__ cosT, float* __restrict__ sinT) {
  int idx = blockIdx.x * blockDim.x + threadIdx.x;   // pos*32 + i
  int pos = idx >> 5, i = idx & 31;
  float inv = __expf(-__logf(10000.0f) * ((float)i / 32.0f));
  float ang = (float)pos * inv;
  float s, c;
  sincosf(ang, &s, &c);
  cosT[idx] = c;
  sinT[idx] = s;
}

// -------------------------------------------------- QKV epilogue (per 16 rows)
static __device__ __forceinline__ void
qkv_store_group(v8f a0, v8f a1, v8f a2, v8f a3, int mrow, int n0,
                int hl, int ln, bool isQ, bool isK,
                const float* __restrict__ cosT, const float* __restrict__ sinT,
                __bf16* __restrict__ Qr, __bf16* __restrict__ Kr,
                __bf16* __restrict__ Vt) {
  #pragma unroll
  for (int g = 0; g < 8; ++g) {
    int row  = mrow + g + hl * 8;
    int pos  = row & (S_LEN - 1);
    int bidx = row >> 11;
    float vF[4] = { a0[g], a1[g], a2[g], a3[g] };
    __bf16 bF[4];
    #pragma unroll
    for (int f = 0; f < 4; ++f) {
      float pv = __shfl_xor(vF[f], 1, 32);   // even/odd dim partner
      float v = vF[f];
      if (isQ || isK) {
        int dc = f * 16 + ln;                // dim within head (n0 64-aligned)
        int fi = dc >> 1;
        float cc = cosT[pos * 32 + fi], sn = sinT[pos * 32 + fi];
        v = (dc & 1) ? (sn * pv + cc * v) : (cc * v - sn * pv);
      }
      bF[f] = (__bf16)v;
    }
    if (isQ) {
      int h = n0 >> 6;
      __bf16* q = Qr + (((size_t)(bidx * NH + h)) * S_LEN + pos) * HD + ln;
      q[0] = bF[0]; q[16] = bF[1]; q[32] = bF[2]; q[48] = bF[3];
    } else if (isK) {
      int h = (n0 - 2048) >> 6;
      __bf16* k = Kr + (((size_t)(bidx * NKV + h)) * S_LEN + pos) * HD + ln;
      k[0] = bF[0]; k[16] = bF[1]; k[32] = bF[2]; k[48] = bF[3];
    } else {
      int h = (n0 - 2560) >> 6;              // V dim-major: [b][kv][dim][key]
      __bf16* v = Vt + ((size_t)(bidx * NKV + h) * HD + ln) * S_LEN + pos;
      v[0]          = bF[0];
      v[16 * S_LEN] = bF[1];
      v[32 * S_LEN] = bF[2];
      v[48 * S_LEN] = bF[3];
    }
  }
}

// --------------------------------------------------------- fused QKV + RoPE
// Block: 8 waves; wave w -> 32-row m-tile x 64-col n-tile (8 accumulators).
// Double-buffered fragment-order LDS B tile; pipelined loads; 1 barrier/step.
__global__ void __launch_bounds__(256)
qkv_rope_kernel(const __bf16* __restrict__ xb,
                const __bf16* __restrict__ WqT, const __bf16* __restrict__ WkT,
                const __bf16* __restrict__ WvT,
                const float* __restrict__ cosT, const float* __restrict__ sinT,
                __bf16* __restrict__ Qr, __bf16* __restrict__ Kr,
                __bf16* __restrict__ Vt) {
  __shared__ __align__(16) __bf16 ldsBT[2][64 * 40];
  const int lane = threadIdx.x & 31, w = threadIdx.x >> 5;
  const int hl = lane >> 4, ln = lane & 15;
  const int m0 = blockIdx.x * 256 + w * 32;
  const int n0 = blockIdx.y * 64;            // one head/region per tile

  const __bf16* WT; int c0;
  if (n0 < 2048)      { WT = WqT; c0 = n0; }
  else if (n0 < 2560) { WT = WkT; c0 = n0 - 2048; }
  else                { WT = WvT; c0 = n0 - 2560; }

  const __bf16* abaseA = xb + (size_t)(m0 + ln) * DM + hl * 8;
  const __bf16* abaseB = abaseA + 16 * DM;
  const int scc = threadIdx.x >> 2;          // staging col 0..63
  const int skk = (threadIdx.x & 3) * 8;     // staging k-group
  const __bf16* sbase = WT + (size_t)(c0 + scc) * DM + skk;
  const int soff = scc * 40 + skk;
  const int foff = ln * 40 + hl * 8;

  v8f accA[4] = {{}, {}, {}, {}};
  v8f accB[4] = {{}, {}, {}, {}};

  // pipeline prologue
  bh8 sreg = *(const bh8*)(sbase);
  BF16x16 aA, aB;
  aA.h[0] = *(const bh8*)(abaseA);      aA.h[1] = *(const bh8*)(abaseA + 16);
  aB.h[0] = *(const bh8*)(abaseB);      aB.h[1] = *(const bh8*)(abaseB + 16);

  for (int k0 = 0; k0 < DM; k0 += 32) {
    const int p = (k0 >> 5) & 1;
    *(bh8*)&ldsBT[p][soff] = sreg;
    __syncthreads();

    // issue next-iteration global loads early (one step past the end on the
    // final iteration reads adjacent, allocated workspace -> harmless)
    sreg = *(const bh8*)(sbase + k0 + 32);
    BF16x16 aAn, aBn;
    aAn.h[0] = *(const bh8*)(abaseA + k0 + 32);
    aAn.h[1] = *(const bh8*)(abaseA + k0 + 48);
    aBn.h[0] = *(const bh8*)(abaseB + k0 + 32);
    aBn.h[1] = *(const bh8*)(abaseB + k0 + 48);

    #pragma unroll
    for (int f = 0; f < 4; ++f) {
      BF16x16 b;
      const __bf16* bp = &ldsBT[p][f * 640 + foff];
      b.h[0] = *(const bh8*)(bp);
      b.h[1] = *(const bh8*)(bp + 16);
      accA[f] = wmma_bf16(aA.v, b.v, accA[f]);
      accB[f] = wmma_bf16(aB.v, b.v, accB[f]);
    }
    aA = aAn; aB = aBn;
  }

  const bool isQ = (n0 < 2048), isK = (n0 >= 2048 && n0 < 2560);
  qkv_store_group(accA[0], accA[1], accA[2], accA[3], m0,      n0, hl, ln,
                  isQ, isK, cosT, sinT, Qr, Kr, Vt);
  qkv_store_group(accB[0], accB[1], accB[2], accB[3], m0 + 16, n0, hl, ln,
                  isQ, isK, cosT, sinT, Qr, Kr, Vt);
}

// ------------------------------------------------------- flash attention
// One wave per (b, h, 16-row q-tile); stream 32-key chunks, online softmax.
// K row-major and V dim-major => all WMMA fragments are contiguous 16B loads.
__global__ void __launch_bounds__(256)
attn_kernel(const __bf16* __restrict__ Qr, const __bf16* __restrict__ Kr,
            const __bf16* __restrict__ Vt, __bf16* __restrict__ AO) {
  __shared__ __align__(16) __bf16 ldsP[8][16 * 40];   // per-wave P staging, padded
  const int lane = threadIdx.x & 31, w = threadIdx.x >> 5;
  const int hl = lane >> 4, ln = lane & 15;
  const int wid  = blockIdx.x * 8 + w;
  const int qt   = wid & 127;
  const int h    = (wid >> 7) & 31;
  const int bidx = wid >> 12;
  const int kvh  = h >> 2;                   // GQA: 4 query heads per KV head
  const int q0   = qt * 16;

  // Q fragments (k-dim = 64 -> two A frags)
  const __bf16* qb = Qr + (((size_t)(bidx * NH + h)) * S_LEN + q0 + ln) * HD + hl * 16;
  BF16x16 aq0, aq1;
  aq0.h[0] = *(const bh8*)(qb);
  aq0.h[1] = *(const bh8*)(qb + 8);
  aq1.h[0] = *(const bh8*)(qb + 32);
  aq1.h[1] = *(const bh8*)(qb + 40);

  const __bf16* kbp = Kr + ((size_t)(bidx * NKV + kvh)) * S_LEN * HD;
  const __bf16* vtp = Vt + ((size_t)(bidx * NKV + kvh)) * HD * S_LEN;

  float m_r[8], l_r[8];
  #pragma unroll
  for (int g = 0; g < 8; ++g) { m_r[g] = -1e30f; l_r[g] = 0.0f; }
  v8f o0 = {}, o1 = {}, o2 = {}, o3 = {};

  const int nchunks = (q0 + 47) >> 5;        // keys 0 .. q0+15, 32 per chunk
  for (int ci = 0; ci < nchunks; ++ci) {
    const int kc = ci * 32;

    // B = K^T fragments: lane = key column, contiguous head-dims per lane
    const __bf16* kb0 = kbp + (size_t)(kc + ln) * HD + hl * 16;
    const __bf16* kb1 = kbp + (size_t)(kc + 16 + ln) * HD + hl * 16;
    BF16x16 b00, b01, b10, b11;
    b00.h[0] = *(const bh8*)(kb0);      b00.h[1] = *(const bh8*)(kb0 + 8);
    b01.h[0] = *(const bh8*)(kb0 + 32); b01.h[1] = *(const bh8*)(kb0 + 40);
    b10.h[0] = *(const bh8*)(kb1);      b10.h[1] = *(const bh8*)(kb1 + 8);
    b11.h[0] = *(const bh8*)(kb1 + 32); b11.h[1] = *(const bh8*)(kb1 + 40);

    // V B-fragments direct from dim-major V: lane = dim column, keys contiguous
    BF16x16 bv0, bv1, bv2, bv3;
    {
      const __bf16* vb = vtp + (size_t)ln * S_LEN + kc + hl * 16;
      bv0.h[0] = *(const bh8*)(vb);
      bv0.h[1] = *(const bh8*)(vb + 8);
      bv1.h[0] = *(const bh8*)(vb + 16 * S_LEN);
      bv1.h[1] = *(const bh8*)(vb + 16 * S_LEN + 8);
      bv2.h[0] = *(const bh8*)(vb + 32 * S_LEN);
      bv2.h[1] = *(const bh8*)(vb + 32 * S_LEN + 8);
      bv3.h[0] = *(const bh8*)(vb + 48 * S_LEN);
      bv3.h[1] = *(const bh8*)(vb + 48 * S_LEN + 8);
    }

    if (ci + 1 < nchunks) {
      __builtin_prefetch(kbp + (size_t)(kc + 32 + ln) * HD, 0, 3);
      __builtin_prefetch(vtp + (size_t)lane * S_LEN + kc + 32, 0, 3);
      __builtin_prefetch(vtp + (size_t)(lane + 32) * S_LEN + kc + 32, 0, 3);
    }

    v8f c0 = {}, c1 = {};                    // scores, two 16-key halves
    c0 = wmma_bf16(aq0.v, b00.v, c0);
    c0 = wmma_bf16(aq1.v, b01.v, c0);
    c1 = wmma_bf16(aq0.v, b10.v, c1);
    c1 = wmma_bf16(aq1.v, b11.v, c1);

    const bool needmask = (kc + 31 > q0);
    float corr[8];
    #pragma unroll
    for (int g = 0; g < 8; ++g) {
      int row = q0 + g + hl * 8;
      float s0 = c0[g] * 0.125f;             // 1/sqrt(64)
      float s1 = c1[g] * 0.125f;
      if (needmask) {
        if (kc + ln > row)      s0 = -1e30f;
        if (kc + 16 + ln > row) s1 = -1e30f;
      }
      // row max across the 16 lanes holding this row (stay in lane-half)
      float mt = fmaxf(s0, s1);
      mt = fmaxf(mt, __shfl_xor(mt, 1, 32));
      mt = fmaxf(mt, __shfl_xor(mt, 2, 32));
      mt = fmaxf(mt, __shfl_xor(mt, 4, 32));
      mt = fmaxf(mt, __shfl_xor(mt, 8, 32));
      float mn = fmaxf(m_r[g], mt);
      float cr = __expf(m_r[g] - mn);
      float e0 = __expf(s0 - mn);
      float e1 = __expf(s1 - mn);
      float ps = e0 + e1;
      ps += __shfl_xor(ps, 1, 32);
      ps += __shfl_xor(ps, 2, 32);
      ps += __shfl_xor(ps, 4, 32);
      ps += __shfl_xor(ps, 8, 32);
      l_r[g] = l_r[g] * cr + ps;
      m_r[g] = mn;
      corr[g] = cr;
      int prow = g + hl * 8;                 // stage P (bf16) for A-frag reload
      ldsP[w][prow * 40 + ln]      = (__bf16)e0;
      ldsP[w][prow * 40 + 16 + ln] = (__bf16)e1;
    }
    #pragma unroll
    for (int g = 0; g < 8; ++g) {
      o0[g] *= corr[g]; o1[g] *= corr[g]; o2[g] *= corr[g]; o3[g] *= corr[g];
    }

    // Wave-private LDS; DS ops are in-order per wave. Fence + clobber keeps
    // the compiler from reordering the P stores vs. the fragment reloads.
    asm volatile("s_wait_dscnt 0" ::: "memory");

    BF16x16 ap;                              // P as A fragment (16x32)
    const __bf16* pb = &ldsP[w][ln * 40 + hl * 16];
    ap.h[0] = *(const bh8*)(pb);
    ap.h[1] = *(const bh8*)(pb + 8);

    o0 = wmma_bf16(ap.v, bv0.v, o0);
    o1 = wmma_bf16(ap.v, bv1.v, o1);
    o2 = wmma_bf16(ap.v, bv2.v, o2);
    o3 = wmma_bf16(ap.v, bv3.v, o3);
  }

  #pragma unroll
  for (int g = 0; g < 8; ++g) {
    float inv = 1.0f / l_r[g];
    int row = q0 + g + hl * 8;
    __bf16* dst = AO + ((size_t)(bidx * S_LEN + row)) * DM + h * HD + ln;
    dst[0]  = (__bf16)(o0[g] * inv);
    dst[16] = (__bf16)(o1[g] * inv);
    dst[32] = (__bf16)(o2[g] * inv);
    dst[48] = (__bf16)(o3[g] * inv);
  }
}

// --------------------------------------------------------- output projection
// Same 32x64-per-wave pipelined structure as QKV; fp32 output.
__global__ void __launch_bounds__(256)
proj_kernel(const __bf16* __restrict__ AO, const __bf16* __restrict__ WoT,
            float* __restrict__ out) {
  __shared__ __align__(16) __bf16 ldsBT[2][64 * 40];
  const int lane = threadIdx.x & 31, w = threadIdx.x >> 5;
  const int hl = lane >> 4, ln = lane & 15;
  const int m0 = blockIdx.x * 256 + w * 32;
  const int n0 = blockIdx.y * 64;
  const __bf16* abaseA = AO + (size_t)(m0 + ln) * DM + hl * 8;
  const __bf16* abaseB = abaseA + 16 * DM;
  const int scc = threadIdx.x >> 2;
  const int skk = (threadIdx.x & 3) * 8;
  const __bf16* sbase = WoT + (size_t)(n0 + scc) * DM + skk;
  const int soff = scc * 40 + skk;
  const int foff = ln * 40 + hl * 8;

  v8f accA[4] = {{}, {}, {}, {}};
  v8f accB[4] = {{}, {}, {}, {}};

  bh8 sreg = *(const bh8*)(sbase);
  BF16x16 aA, aB;
  aA.h[0] = *(const bh8*)(abaseA);      aA.h[1] = *(const bh8*)(abaseA + 16);
  aB.h[0] = *(const bh8*)(abaseB);      aB.h[1] = *(const bh8*)(abaseB + 16);

  for (int k0 = 0; k0 < DM; k0 += 32) {
    const int p = (k0 >> 5) & 1;
    *(bh8*)&ldsBT[p][soff] = sreg;
    __syncthreads();

    sreg = *(const bh8*)(sbase + k0 + 32);   // one step OOB at end: safe in ws
    BF16x16 aAn, aBn;
    aAn.h[0] = *(const bh8*)(abaseA + k0 + 32);
    aAn.h[1] = *(const bh8*)(abaseA + k0 + 48);
    aBn.h[0] = *(const bh8*)(abaseB + k0 + 32);
    aBn.h[1] = *(const bh8*)(abaseB + k0 + 48);

    #pragma unroll
    for (int f = 0; f < 4; ++f) {
      BF16x16 b;
      const __bf16* bp = &ldsBT[p][f * 640 + foff];
      b.h[0] = *(const bh8*)(bp);
      b.h[1] = *(const bh8*)(bp + 16);
      accA[f] = wmma_bf16(aA.v, b.v, accA[f]);
      accB[f] = wmma_bf16(aB.v, b.v, accB[f]);
    }
    aA = aAn; aB = aBn;
  }

  #pragma unroll
  for (int g = 0; g < 8; ++g) {
    size_t rowA = (size_t)(m0 + g + hl * 8);
    float* oA = out + rowA * DM + n0 + ln;
    oA[0] = accA[0][g]; oA[16] = accA[1][g];
    oA[32] = accA[2][g]; oA[48] = accA[3][g];
    float* oB = oA + 16 * DM;
    oB[0] = accB[0][g]; oB[16] = accB[1][g];
    oB[32] = accB[2][g]; oB[48] = accB[3][g];
  }
}

// --------------------------------------------------------------------------
extern "C" void kernel_launch(void* const* d_in, const int* in_sizes, int n_in,
                              void* d_out, int out_size, void* d_ws, size_t ws_size,
                              hipStream_t stream) {
  (void)in_sizes; (void)n_in; (void)out_size; (void)ws_size;
  const float* x  = (const float*)d_in[0];
  const float* Wq = (const float*)d_in[1];
  const float* Wk = (const float*)d_in[2];
  const float* Wv = (const float*)d_in[3];
  const float* Wo = (const float*)d_in[4];
  float* out = (float*)d_out;
  char* ws = (char*)d_ws;

  // workspace layout (bytes), total ~80.2 MB
  const size_t OFF_WQ  = 0;                    // WqT: 2048x2048 bf16
  const size_t OFF_WK  = OFF_WQ  + 8388608;    // WkT: 512x2048  bf16
  const size_t OFF_WV  = OFF_WK  + 2097152;    // WvT: 512x2048  bf16
  const size_t OFF_WO  = OFF_WV  + 2097152;    // WoT: 2048x2048 bf16
  const size_t OFF_COS = OFF_WO  + 8388608;    // 2048*32 f32
  const size_t OFF_SIN = OFF_COS + 262144;
  const size_t OFF_XB  = OFF_SIN + 262144;     // x bf16: 4096x2048
  const size_t OFF_Q   = OFF_XB  + 16777216;   // 2*32*2048*64 bf16
  const size_t OFF_K   = OFF_Q   + 16777216;   // 2*8*2048*64  bf16
  const size_t OFF_V   = OFF_K   + 4194304;    // 2*8*64*2048  bf16 (dim-major)
  const size_t OFF_AO  = OFF_V   + 4194304;    // 2*2048*2048  bf16

  __bf16* wqT = (__bf16*)(ws + OFF_WQ);
  __bf16* wkT = (__bf16*)(ws + OFF_WK);
  __bf16* wvT = (__bf16*)(ws + OFF_WV);
  __bf16* woT = (__bf16*)(ws + OFF_WO);
  float*  cosT = (float*)(ws + OFF_COS);
  float*  sinT = (float*)(ws + OFF_SIN);
  __bf16* xb  = (__bf16*)(ws + OFF_XB);
  __bf16* Qr  = (__bf16*)(ws + OFF_Q);
  __bf16* Kr  = (__bf16*)(ws + OFF_K);
  __bf16* Vt  = (__bf16*)(ws + OFF_V);
  __bf16* AO  = (__bf16*)(ws + OFF_AO);

  // one-time transpose-convert of weights (coalesced both sides)
  transpose_cvt<<<dim3(64, 64), 256, 0, stream>>>(Wq, wqT, 2048);
  transpose_cvt<<<dim3(16, 64), 256, 0, stream>>>(Wk, wkT, 512);
  transpose_cvt<<<dim3(16, 64), 256, 0, stream>>>(Wv, wvT, 512);
  transpose_cvt<<<dim3(64, 64), 256, 0, stream>>>(Wo, woT, 2048);
  // one-time x fp32 -> bf16
  cvt_f32_bf16<<<(8388608 + 255) / 256, 256, 0, stream>>>(x, xb, 8388608);
  rope_table<<<65536 / 256, 256, 0, stream>>>(cosT, sinT);

  // fused [Q|K|V] projection + RoPE: M=4096 rows, N=3072 cols (32x64/wave)
  qkv_rope_kernel<<<dim3(16, 48), 256, 0, stream>>>(xb, wqT, wkT, wvT,
                                                    cosT, sinT, Qr, Kr, Vt);
  // flash attention: 2*32*128 q-tiles = 8192 waves, 8 waves/block
  attn_kernel<<<1024, 256, 0, stream>>>(Qr, Kr, Vt, AO);
  // output projection: (4096 x 2048) @ (2048 x 2048) -> fp32
  proj_kernel<<<dim3(16, 32), 256, 0, stream>>>(AO, woT, out);
}
